// Attention_12146167513711
// MI455X (gfx1250) — compile-verified, matrix operand order checked by reading, not run on hardware
//
#include <hip/hip_runtime.h>
#include <hip/hip_bf16.h>
#include <math.h>

// ---------------- model dims ----------------
#define S_SRC 48
#define T_TGT 32
#define BB    32
#define DD    512
#define NHH   8
#define HDD   64
#define FFF   2048
#define VTT   32000

typedef __attribute__((ext_vector_type(16))) __bf16 v16bf;
typedef __attribute__((ext_vector_type(8)))  float  v8f;

__device__ __forceinline__ unsigned short f2bf(float f) {
  unsigned int x = __float_as_uint(f);
  unsigned int r = x + 0x7FFFu + ((x >> 16) & 1u);   // round-to-nearest-even
  return (unsigned short)(r >> 16);
}

__device__ __forceinline__ float sigmoidf_(float x) { return 1.0f / (1.0f + __expf(-x)); }

// Load one 16x32 bf16 fragment slice for this lane from a row-major f32 row.
// ISA bf16 A/B fragment: lane L holds K = {koff..koff+7} and {16+koff..16+koff+7},
// koff = (L>=16)*8  -> two contiguous 8-float runs -> 4x b128 loads + pack.
__device__ __forceinline__ v16bf load_frag(const float* __restrict__ rowp, int k0, int koff) {
  const float4* p = (const float4*)(rowp + k0 + koff);
  const float4* q = (const float4*)(rowp + k0 + 16 + koff);
  float4 a0 = p[0], a1 = p[1], a2 = q[0], a3 = q[1];
  union { v16bf v; unsigned short u[16]; } f;
  f.u[0]=f2bf(a0.x);  f.u[1]=f2bf(a0.y);  f.u[2] =f2bf(a0.z);  f.u[3] =f2bf(a0.w);
  f.u[4]=f2bf(a1.x);  f.u[5]=f2bf(a1.y);  f.u[6] =f2bf(a1.z);  f.u[7] =f2bf(a1.w);
  f.u[8]=f2bf(a2.x);  f.u[9]=f2bf(a2.y);  f.u[10]=f2bf(a2.z);  f.u[11]=f2bf(a2.w);
  f.u[12]=f2bf(a3.x); f.u[13]=f2bf(a3.y); f.u[14]=f2bf(a3.z);  f.u[15]=f2bf(a3.w);
  return f.v;
}

// ---------------------------------------------------------------------------
// C[M,N] = act( A[M,K] @ W[N,K]^T + bias[N] + res[M,N] )
// bf16 WMMA, f32 accumulate. M%32==0, K%32==0, N%64==0 for active waves.
// One wave -> 32x64 C tile (2x4 fragments, 8 WMMAs per k-step).
// Block = 8 waves -> 32x512 tile.
// ---------------------------------------------------------------------------
__global__ void __launch_bounds__(256)
gemm_bf16_wmma(const float* __restrict__ A, const float* __restrict__ W,
               const float* __restrict__ bias, const float* __restrict__ res,
               float* __restrict__ C, int M, int N, int K, int relu)
{
  const int lane = threadIdx.x & 31;
  const int wave = threadIdx.x >> 5;                 // 0..7
  const int m0 = blockIdx.y * 32;
  const int n0 = blockIdx.x * 512 + wave * 64;
  if (m0 >= M || n0 >= N) return;                    // wave-uniform: EXEC stays all-1s

  const int row  = lane & 15;                        // M index (A) / N index (W)
  const int koff = (lane >> 4) * 8;                  // per ISA bf16 fragment layout

  const float* __restrict__ arow[2];
  arow[0] = A + (size_t)(m0 + row) * K;
  arow[1] = A + (size_t)(m0 + 16 + row) * K;
  const float* __restrict__ wrow[4];
#pragma unroll
  for (int j = 0; j < 4; ++j) wrow[j] = W + (size_t)(n0 + j * 16 + row) * K;

  v8f acc[2][4] = {};

  for (int k0 = 0; k0 < K; k0 += 32) {
    if (k0 + 64 < K) {   // prefetch 2 k-steps ahead, WGP scope (fills all cache levels)
#pragma unroll
      for (int i = 0; i < 2; ++i) __builtin_prefetch(arow[i] + k0 + 64 + koff, 0, 3);
#pragma unroll
      for (int j = 0; j < 4; ++j) __builtin_prefetch(wrow[j] + k0 + 64 + koff, 0, 3);
    }
    v16bf av[2], bv[4];
#pragma unroll
    for (int i = 0; i < 2; ++i) av[i] = load_frag(arow[i], k0, koff);
#pragma unroll
    for (int j = 0; j < 4; ++j) bv[j] = load_frag(wrow[j], k0, koff);
#pragma unroll
    for (int i = 0; i < 2; ++i)
#pragma unroll
      for (int j = 0; j < 4; ++j)
        acc[i][j] = __builtin_amdgcn_wmma_f32_16x16x32_bf16(false, av[i], false, bv[j],
                                                            (short)0, acc[i][j], false, false);
  }

  // C/D fragment: VGPR i -> M = (lane>=16?8:0)+i, N = lane&15
#pragma unroll
  for (int mi = 0; mi < 2; ++mi) {
    const int cmb = m0 + mi * 16 + (lane >> 4) * 8;
#pragma unroll
    for (int nj = 0; nj < 4; ++nj) {
      const int cn = n0 + nj * 16 + (lane & 15);
      const float bval = bias ? bias[cn] : 0.0f;
#pragma unroll
      for (int i = 0; i < 8; ++i) {
        const int cm = cmb + i;
        float v = acc[mi][nj][i] + bval;
        if (res)  v += res[(size_t)cm * N + cn];
        if (relu) v = fmaxf(v, 0.0f);
        C[(size_t)cm * N + cn] = v;
      }
    }
  }
}

// ---------------------------------------------------------------------------
// Embedding + sinusoidal positional encoding: out[s,b,d] = emb[id[s,b],d]+pe(s,d)
// ---------------------------------------------------------------------------
__global__ void embed_pe_kernel(const int* __restrict__ ids, const float* __restrict__ emb,
                                float* __restrict__ out, int S, int Bn, int Dn)
{
  int idx = blockIdx.x * blockDim.x + threadIdx.x;
  int total = S * Bn * Dn;
  if (idx >= total) return;
  int d = idx % Dn;
  int b = (idx / Dn) % Bn;
  int s = idx / (Dn * Bn);
  int pair = d >> 1;
  float div = __expf((2.0f * (float)pair) * (-logf(10000.0f) / (float)Dn));
  float ang = (float)s * div;
  float pe  = (d & 1) ? __cosf(ang) : __sinf(ang);
  out[idx] = emb[(size_t)ids[s * Bn + b] * Dn + d] + pe;
}

// ---------------------------------------------------------------------------
// Encoder self-attention (small: VALU path)
// ---------------------------------------------------------------------------
__global__ void attn_scores_kernel(const float* __restrict__ qkv, float* __restrict__ sc)
{
  int idx = blockIdx.x * blockDim.x + threadIdx.x;
  int total = BB * NHH * S_SRC * S_SRC;
  if (idx >= total) return;
  int t = idx % S_SRC;
  int s = (idx / S_SRC) % S_SRC;
  int h = (idx / (S_SRC * S_SRC)) % NHH;
  int b = idx / (S_SRC * S_SRC * NHH);
  const float* q = qkv + ((size_t)(s * BB + b)) * 3 * DD + h * HDD;
  const float* k = qkv + ((size_t)(t * BB + b)) * 3 * DD + DD + h * HDD;
  float acc = 0.0f;
#pragma unroll 8
  for (int d = 0; d < HDD; ++d) acc += q[d] * k[d];
  sc[idx] = acc * 0.125f;                            // 1/sqrt(64)
}

__global__ void row_softmax_kernel(float* __restrict__ x, int rows, int n)
{
  int r = blockIdx.x * blockDim.x + threadIdx.x;
  if (r >= rows) return;
  float* xr = x + (size_t)r * n;
  float mx = -INFINITY;
  for (int i = 0; i < n; ++i) mx = fmaxf(mx, xr[i]);
  float sum = 0.0f;
  for (int i = 0; i < n; ++i) { float e = __expf(xr[i] - mx); xr[i] = e; sum += e; }
  float inv = 1.0f / sum;
  for (int i = 0; i < n; ++i) xr[i] *= inv;
}

__global__ void attn_av_kernel(const float* __restrict__ a, const float* __restrict__ qkv,
                               float* __restrict__ o)
{
  int idx = blockIdx.x * blockDim.x + threadIdx.x;
  int total = S_SRC * BB * DD;
  if (idx >= total) return;
  int d = idx % DD;
  int b = (idx / DD) % BB;
  int s = idx / (DD * BB);
  int h = d / HDD;
  const float* ar = a + (((size_t)(b * NHH + h) * S_SRC) + s) * S_SRC;
  float acc = 0.0f;
  for (int t = 0; t < S_SRC; ++t)
    acc += ar[t] * qkv[((size_t)(t * BB + b)) * 3 * DD + 2 * DD + d];
  o[idx] = acc;
}

// ---------------------------------------------------------------------------
// LayerNorm (block per row) and log-softmax (block per row)
// ---------------------------------------------------------------------------
__global__ void layernorm_kernel(float* __restrict__ x, const float* __restrict__ w,
                                 const float* __restrict__ b, int Dn)
{
  __shared__ float red[256];
  float* xr = x + (size_t)blockIdx.x * Dn;
  int tid = threadIdx.x;
  float s = 0.0f;
  for (int i = tid; i < Dn; i += 256) s += xr[i];
  red[tid] = s; __syncthreads();
  for (int o = 128; o > 0; o >>= 1) { if (tid < o) red[tid] += red[tid + o]; __syncthreads(); }
  float mean = red[0] / (float)Dn; __syncthreads();
  float v = 0.0f;
  for (int i = tid; i < Dn; i += 256) { float t = xr[i] - mean; v += t * t; }
  red[tid] = v; __syncthreads();
  for (int o = 128; o > 0; o >>= 1) { if (tid < o) red[tid] += red[tid + o]; __syncthreads(); }
  float rstd = rsqrtf(red[0] / (float)Dn + 1e-5f); __syncthreads();
  for (int i = tid; i < Dn; i += 256) xr[i] = (xr[i] - mean) * rstd * w[i] + b[i];
}

__global__ void log_softmax_kernel(float* __restrict__ x, int n)
{
  __shared__ float red[256];
  float* xr = x + (size_t)blockIdx.x * n;
  int tid = threadIdx.x;
  float mx = -INFINITY;
  for (int i = tid; i < n; i += 256) mx = fmaxf(mx, xr[i]);
  red[tid] = mx; __syncthreads();
  for (int o = 128; o > 0; o >>= 1) { if (tid < o) red[tid] = fmaxf(red[tid], red[tid + o]); __syncthreads(); }
  mx = red[0]; __syncthreads();
  float s = 0.0f;
  for (int i = tid; i < n; i += 256) s += __expf(xr[i] - mx);
  red[tid] = s; __syncthreads();
  for (int o = 128; o > 0; o >>= 1) { if (tid < o) red[tid] += red[tid + o]; __syncthreads(); }
  float lse = mx + __logf(red[0]); __syncthreads();
  for (int i = tid; i < n; i += 256) xr[i] -= lse;
}

// ---------------------------------------------------------------------------
// LSTM cell: gates g[B,4D] order (i,f,g,o); writes h into concat slot.
// ---------------------------------------------------------------------------
__global__ void lstm_cell_kernel(const float* __restrict__ g, float* __restrict__ h,
                                 float* __restrict__ c, float* __restrict__ cat)
{
  int idx = blockIdx.x * blockDim.x + threadIdx.x;
  if (idx >= BB * DD) return;
  int b = idx / DD, d = idx % DD;
  const float* gb = g + (size_t)b * 4 * DD;
  float ig = sigmoidf_(gb[d]);
  float fg = sigmoidf_(gb[DD + d]);
  float gg = tanhf(gb[2 * DD + d]);
  float og = sigmoidf_(gb[3 * DD + d]);
  float cn = fg * c[idx] + ig * gg;
  float hn = og * tanhf(cn);
  c[idx] = cn; h[idx] = hn;
  cat[(size_t)b * 2 * DD + d] = hn;
}

// scores[s,b] = h[b,:] . p[s,b,:]
__global__ void attend_scores_kernel(const float* __restrict__ h, const float* __restrict__ p,
                                     float* __restrict__ sc)
{
  int idx = blockIdx.x * blockDim.x + threadIdx.x;
  if (idx >= S_SRC * BB) return;
  int b = idx % BB, s = idx / BB;
  const float* hb = h + (size_t)b * DD;
  const float* ps = p + ((size_t)(s * BB + b)) * DD;
  float acc = 0.0f;
  for (int d = 0; d < DD; ++d) acc += hb[d] * ps[d];
  sc[idx] = acc;
}

// softmax over s (axis 0) per column b, on sc stored [S,B] flat
__global__ void attend_colsoftmax_kernel(float* __restrict__ sc)
{
  int b = threadIdx.x;
  if (b >= BB) return;
  float mx = -INFINITY;
  for (int s = 0; s < S_SRC; ++s) mx = fmaxf(mx, sc[s * BB + b]);
  float sum = 0.0f;
  for (int s = 0; s < S_SRC; ++s) sum += __expf(sc[s * BB + b] - mx);
  float inv = 1.0f / sum;
  for (int s = 0; s < S_SRC; ++s) sc[s * BB + b] = __expf(sc[s * BB + b] - mx) * inv;
}

// ctx[b,d] = sum_s aw[b,s]*enc[s,b,d], with faithful buggy flat reshape:
// aw[b,s] = sc_flat[b*S + s]  (sc is [S,B] flat)
__global__ void attend_ctx_kernel(const float* __restrict__ sc, const float* __restrict__ enc,
                                  float* __restrict__ out, int ostride)
{
  int idx = blockIdx.x * blockDim.x + threadIdx.x;
  if (idx >= BB * DD) return;
  int b = idx / DD, d = idx % DD;
  float acc = 0.0f;
  for (int s = 0; s < S_SRC; ++s)
    acc += sc[b * S_SRC + s] * enc[((size_t)(s * BB + b)) * DD + d];
  out[(size_t)b * ostride + d] = acc;
}

// ---------------------------------------------------------------------------
extern "C" void kernel_launch(void* const* d_in, const int* in_sizes, int n_in,
                              void* d_out, int out_size, void* d_ws, size_t ws_size,
                              hipStream_t stream)
{
  (void)in_sizes; (void)n_in; (void)out_size; (void)ws_size;
  // input indices (setup_inputs dict order; nested dicts flatten in key order)
  const int*   src    = (const int*)d_in[0];
  const int*   tgt    = (const int*)d_in[1];
  const float* h0     = (const float*)d_in[2];
  const float* c0     = (const float*)d_in[3];
  const float* eemb   = (const float*)d_in[4];
  const float* demb   = (const float*)d_in[5];
  const float* const* enc1 = (const float* const*)&d_in[6];   // 12 params
  const float* const* enc2 = (const float* const*)&d_in[18];  // 12 params
  const float* const* lst1 = (const float* const*)&d_in[30];  // w_ih,w_hh,b_ih,b_hh
  const float* const* lst2 = (const float* const*)&d_in[34];
  const float* attn_w = (const float*)d_in[38];
  const float* attn_b = (const float*)d_in[39];
  const float* out_w  = (const float*)d_in[40];
  const float* out_b  = (const float*)d_in[41];
  float* out = (float*)d_out;

  // workspace bump allocator (floats, 64-elt aligned)
  float* ws = (float*)d_ws;
  size_t off = 0;
  auto alloc = [&](size_t n) { float* p = ws + off; off += (n + 63) & ~(size_t)63; return p; };
  const int MT = S_SRC * BB;   // 1536 encoder tokens
  float* xbuf   = alloc((size_t)MT * DD);
  float* qkv    = alloc((size_t)MT * 3 * DD);
  float* scores = alloc((size_t)BB * NHH * S_SRC * S_SRC);
  float* aobuf  = alloc((size_t)MT * DD);
  float* x2buf  = alloc((size_t)MT * DD);
  float* ffbuf  = alloc((size_t)MT * FFF);
  float* e1out  = alloc((size_t)MT * DD);
  float* e2out  = alloc((size_t)MT * DD);
  float* p1     = alloc((size_t)MT * DD);
  float* p2     = alloc((size_t)MT * DD);
  float* decin  = alloc((size_t)T_TGT * BB * DD);
  float* hbuf   = alloc((size_t)BB * DD);
  float* cbuf   = alloc((size_t)BB * DD);
  float* gbuf   = alloc((size_t)BB * 4 * DD);
  float* attsc  = alloc((size_t)S_SRC * BB);
  float* layer1 = alloc((size_t)T_TGT * BB * 2 * DD);
  float* y2     = alloc((size_t)T_TGT * BB * 2 * DD);

  auto gemm = [&](const float* A, const float* W, const float* bias, const float* res,
                  float* C, int M, int N, int K, int relu) {
    dim3 grid((N + 511) / 512, M / 32);
    gemm_bf16_wmma<<<grid, 256, 0, stream>>>(A, W, bias, res, C, M, N, K, relu);
  };

  auto enc_layer = [&](const float* xin, float* xout, const float* const* p) {
    // p: in_w,in_b,out_w,out_b,l1_w,l1_b,l2_w,l2_b,ln1_w,ln1_b,ln2_w,ln2_b
    gemm(xin, p[0], p[1], nullptr, qkv, MT, 3 * DD, DD, 0);
    int nsc = BB * NHH * S_SRC * S_SRC;
    attn_scores_kernel<<<(nsc + 255) / 256, 256, 0, stream>>>(qkv, scores);
    int nrows = BB * NHH * S_SRC;
    row_softmax_kernel<<<(nrows + 255) / 256, 256, 0, stream>>>(scores, nrows, S_SRC);
    int nav = MT * DD;
    attn_av_kernel<<<(nav + 255) / 256, 256, 0, stream>>>(scores, qkv, aobuf);
    gemm(aobuf, p[2], p[3], xin, x2buf, MT, DD, DD, 0);
    layernorm_kernel<<<MT, 256, 0, stream>>>(x2buf, p[8], p[9], DD);
    gemm(x2buf, p[4], p[5], nullptr, ffbuf, MT, FFF, DD, 1);
    gemm(ffbuf, p[6], p[7], x2buf, xout, MT, DD, FFF, 0);
    layernorm_kernel<<<MT, 256, 0, stream>>>(xout, p[10], p[11], DD);
  };

  // ---------------- encoder ----------------
  int nemb = MT * DD;
  embed_pe_kernel<<<(nemb + 255) / 256, 256, 0, stream>>>(src, eemb, xbuf, S_SRC, BB, DD);
  enc_layer(xbuf, e1out, enc1);
  enc_layer(e1out, e2out, enc2);

  // attention projections of encoder outputs
  gemm(e1out, attn_w, attn_b, nullptr, p1, MT, DD, DD, 0);
  gemm(e2out, attn_w, attn_b, nullptr, p2, MT, DD, DD, 0);

  // ---------------- decoder input ----------------
  int ndec = T_TGT * BB * DD;
  embed_pe_kernel<<<(ndec + 255) / 256, 256, 0, stream>>>(tgt, demb, decin, T_TGT, BB, DD);

  auto decoder = [&](const float* xseq, int xdim, const float* const* lp,
                     const float* enc, const float* pproj, float* catout) {
    hipMemcpyAsync(hbuf, h0, (size_t)BB * DD * sizeof(float), hipMemcpyDeviceToDevice, stream);
    hipMemcpyAsync(cbuf, c0, (size_t)BB * DD * sizeof(float), hipMemcpyDeviceToDevice, stream);
    for (int t = 0; t < T_TGT; ++t) {
      const float* xt = xseq + (size_t)t * BB * xdim;
      gemm(xt,   lp[0], lp[2], nullptr, gbuf, BB, 4 * DD, xdim, 0);   // x@w_ih^T + b_ih
      gemm(hbuf, lp[1], lp[3], gbuf,    gbuf, BB, 4 * DD, DD,   0);   // += h@w_hh^T + b_hh
      float* cat = catout + (size_t)t * BB * 2 * DD;
      lstm_cell_kernel<<<(BB * DD + 255) / 256, 256, 0, stream>>>(gbuf, hbuf, cbuf, cat);
      attend_scores_kernel<<<(S_SRC * BB + 255) / 256, 256, 0, stream>>>(hbuf, pproj, attsc);
      attend_colsoftmax_kernel<<<1, BB, 0, stream>>>(attsc);
      attend_ctx_kernel<<<(BB * DD + 255) / 256, 256, 0, stream>>>(attsc, enc, cat + DD, 2 * DD);
    }
  };

  decoder(decin,  DD,     lst1, e1out, p1, layer1);
  decoder(layer1, 2 * DD, lst2, e2out, p2, y2);

  // ---------------- final projection + log-softmax (dominant GEMM) ----------
  gemm(y2, out_w, out_b, nullptr, out, T_TGT * BB, VTT, 2 * DD, 0);
  log_softmax_kernel<<<T_TGT * BB, 256, 0, stream>>>(out, VTT);
}
